// SgpiSTFT_33509334844088
// MI455X (gfx1250) — matrix operand order repaired and checked by reading, not compiled, not run on hardware
//
#include <hip/hip_runtime.h>
#include <hip/hip_bf16.h>

// ISTFT-like op reduced to: per-batch Y(512x4000) = W(512x512) @ X(512x4000),
// then windowed 4-tap overlap-add epilogue. GEMM uses split-bf16 WMMA
// (v_wmma_f32_16x16x32_bf16) with fp32 accumulation: Wh*Xh + Wl*Xh + Wh*Xl.

typedef __attribute__((ext_vector_type(16))) __bf16 v16bf;
typedef __attribute__((ext_vector_type(8)))  float  v8f;

#define NROWS 512      // M: output rows of Y
#define KDIM  512      // K: used input rows
#define TDIM  4000     // N: time columns
#define TOUT  3999
#define BATCH 8

#define BM 128
#define BN 64
#define BK 32
#define ASTR 40        // LDS row stride in halfwords (multiple of 8 for b128)
#define XSTR 40

union Frag {
    uint4 q[2];
    v16bf v;
};

// ---------------------------------------------------------------- W builder
__global__ __launch_bounds__(256) void build_w_kernel(unsigned short* __restrict__ Whi,
                                                      unsigned short* __restrict__ Wlo) {
    int idx = blockIdx.x * 256 + threadIdx.x;   // 0 .. 512*512-1
    int n = idx >> 9;          // output row
    int j = idx & 511;         // gemm K column
    int jj = (j < 256) ? j : (j - 256);
    int p = (jj * n) & 511;    // exact integer phase reduction mod 512
    float ang = (float)p * (2.0f * 3.14159265358979323846f / 512.0f);
    float w;
    if (j < 256) {
        float a = (jj == 0) ? 1.0f : 2.0f;
        w = a * cosf(ang);
    } else {
        w = (jj == 0) ? 0.0f : 2.0f * sinf(ang);
    }
    __bf16 hi = (__bf16)w;
    float rem = w - (float)hi;
    __bf16 lo = (__bf16)rem;
    Whi[idx] = __builtin_bit_cast(unsigned short, hi);
    Wlo[idx] = __builtin_bit_cast(unsigned short, lo);
}

// ---------------------------------------------------------------- GEMM
// Y[b][m][t] = sum_j W[m][j] * X[b][jmap(j)][t],  jmap(j) = j + (j>=256)
__global__ __launch_bounds__(256) void gemm_kernel(const float* __restrict__ X,
                                                   const unsigned short* __restrict__ Whi,
                                                   const unsigned short* __restrict__ Wlo,
                                                   float* __restrict__ Y) {
    __shared__ __align__(16) unsigned short sAhi[BM * ASTR];
    __shared__ __align__(16) unsigned short sAlo[BM * ASTR];
    __shared__ __align__(16) unsigned short sXhi[BN * XSTR];
    __shared__ __align__(16) unsigned short sXlo[BN * XSTR];

    const int tid  = threadIdx.x;
    const int lane = tid & 31;
    const int wave = tid >> 5;          // 8 waves
    const int wm   = wave >> 1;         // 0..3 -> 32-row slab of M
    const int wn   = wave & 1;          // 0..1 -> 32-col slab of N
    const int half = lane >> 4;
    const int l16  = lane & 15;

    const int b  = blockIdx.z;
    const int n0 = blockIdx.x * BN;
    const int m0 = blockIdx.y * BM;

    const float* Xb = X + (size_t)b * 514 * TDIM;
    float*       Yb = Y + (size_t)b * NROWS * TDIM;

    v8f acc[2][2] = {};

    // W-tile loader mapping: 256 thr -> 128 rows x 2 16B segments
    const int arow = tid >> 1;
    const int aseg = tid & 1;
    // X-tile loader mapping: 256 thr -> 32 k-rows x 8 float4-pairs
    const int krow = tid >> 3;
    const int t8   = (tid & 7) * 8;

    for (int k0 = 0; k0 < KDIM; k0 += BK) {
        // ---- stage W (hi & lo) tile: 128 x 32 bf16, row-major with pad
        {
            const uint4* gh = reinterpret_cast<const uint4*>(Whi + (size_t)(m0 + arow) * KDIM + k0) + aseg;
            const uint4* gl = reinterpret_cast<const uint4*>(Wlo + (size_t)(m0 + arow) * KDIM + k0) + aseg;
            *reinterpret_cast<uint4*>(&sAhi[arow * ASTR + aseg * 8]) = gh[0];
            *reinterpret_cast<uint4*>(&sAlo[arow * ASTR + aseg * 8]) = gl[0];
            *reinterpret_cast<uint4*>(&sAhi[arow * ASTR + aseg * 8 + 16]) = gh[2];
            *reinterpret_cast<uint4*>(&sAlo[arow * ASTR + aseg * 8 + 16]) = gl[2];
        }
        // ---- stage X tile: load f32 (coalesced along t), split hi/lo bf16,
        //      store transposed [t_local][k] for contiguous B-fragments
        {
            int j = k0 + krow;
            int jmap = j + (j >= 256);
            const float* src = Xb + (size_t)jmap * TDIM;
#pragma unroll
            for (int q = 0; q < 2; ++q) {
                int tlocal = t8 + q * 4;
                int tglob  = n0 + tlocal;
                if (tglob > TDIM - 4) tglob = TDIM - 4;   // clamp, junk cols dropped
                float4 v = *reinterpret_cast<const float4*>(src + tglob);
                float vv[4] = {v.x, v.y, v.z, v.w};
#pragma unroll
                for (int e = 0; e < 4; ++e) {
                    float f = vv[e];
                    __bf16 h = (__bf16)f;
                    float rem = f - (float)h;
                    __bf16 l = (__bf16)rem;
                    int off = (tlocal + e) * XSTR + krow;
                    sXhi[off] = __builtin_bit_cast(unsigned short, h);
                    sXlo[off] = __builtin_bit_cast(unsigned short, l);
                }
            }
        }
        __syncthreads();

        // ---- fragments (two contiguous b128 chunks per 16x32 bf16 operand)
        Frag ah[2], al[2], bh[2], bl[2];
#pragma unroll
        for (int mi = 0; mi < 2; ++mi) {
            int ml = wm * 32 + mi * 16 + l16;
            const uint4* ph = reinterpret_cast<const uint4*>(&sAhi[ml * ASTR + half * 8]);
            const uint4* pl = reinterpret_cast<const uint4*>(&sAlo[ml * ASTR + half * 8]);
            ah[mi].q[0] = ph[0]; ah[mi].q[1] = ph[2];
            al[mi].q[0] = pl[0]; al[mi].q[1] = pl[2];
        }
#pragma unroll
        for (int ni = 0; ni < 2; ++ni) {
            int nl = wn * 32 + ni * 16 + l16;
            const uint4* ph = reinterpret_cast<const uint4*>(&sXhi[nl * XSTR + half * 8]);
            const uint4* pl = reinterpret_cast<const uint4*>(&sXlo[nl * XSTR + half * 8]);
            bh[ni].q[0] = ph[0]; bh[ni].q[1] = ph[2];
            bl[ni].q[0] = pl[0]; bl[ni].q[1] = pl[2];
        }

        // ---- 3-product split-bf16 accumulation
#pragma unroll
        for (int mi = 0; mi < 2; ++mi) {
#pragma unroll
            for (int ni = 0; ni < 2; ++ni) {
                v8f a = acc[mi][ni];
                a = __builtin_amdgcn_wmma_f32_16x16x32_bf16(false, ah[mi].v, false, bh[ni].v,
                                                            (short)0, a, false, false);
                a = __builtin_amdgcn_wmma_f32_16x16x32_bf16(false, al[mi].v, false, bh[ni].v,
                                                            (short)0, a, false, false);
                a = __builtin_amdgcn_wmma_f32_16x16x32_bf16(false, ah[mi].v, false, bl[ni].v,
                                                            (short)0, a, false, false);
                acc[mi][ni] = a;
            }
        }
        __syncthreads();
    }

    // ---- store Y (16x16 f32 C layout: VGPR i -> M = half*8 + i, N = l16)
#pragma unroll
    for (int mi = 0; mi < 2; ++mi) {
#pragma unroll
        for (int ni = 0; ni < 2; ++ni) {
            int t = n0 + wn * 32 + ni * 16 + l16;
            if (t < TDIM) {
                int mbase = m0 + wm * 32 + mi * 16 + half * 8;
#pragma unroll
                for (int i = 0; i < 8; ++i)
                    Yb[(size_t)(mbase + i) * TDIM + t] = acc[mi][ni][i];
            }
        }
    }
}

// ---------------------------------------------------------------- epilogue
// out[b,t,k] = 256*( win[1023-k]*Y[511-k,t] + [t>=1] win[767-k]*Y[255-k,t-1]
//                  + [t>=2] win[511-k]*Y[511-k,t-2] + [t>=3] win[255-k]*Y[255-k,t-3] )
__global__ __launch_bounds__(256) void epilogue_kernel(const float* __restrict__ Y,
                                                       const float* __restrict__ win,
                                                       float* __restrict__ out) {
    __shared__ float tile[32][33];
    const int b  = blockIdx.z;
    const int k0 = blockIdx.y * 32;
    const int t0 = blockIdx.x * 32;
    const float* Yb = Y + (size_t)b * NROWS * TDIM;

    const int tid = threadIdx.x;
    {   // phase 1: t-fast mapping -> coalesced Y reads
        const int tt  = tid & 31;
        const int kk8 = tid >> 5;
#pragma unroll
        for (int r = 0; r < 4; ++r) {
            int kl = r * 8 + kk8;
            int k  = k0 + kl;
            int t  = t0 + tt;
            float v = 0.0f;
            if (t < TOUT) {
                const float* ya = Yb + (size_t)(511 - k) * TDIM;
                const float* yc = Yb + (size_t)(255 - k) * TDIM;
                v = win[1023 - k] * ya[t];
                if (t >= 1) v += win[767 - k] * yc[t - 1];
                if (t >= 2) v += win[511 - k] * ya[t - 2];
                if (t >= 3) v += win[255 - k] * yc[t - 3];
                v *= 256.0f;
            }
            tile[tt][kl] = v;
        }
    }
    __syncthreads();
    {   // phase 2: k-fast mapping -> coalesced out writes
        const int kl = tid & 31;
        const int tb = tid >> 5;
#pragma unroll
        for (int r = 0; r < 4; ++r) {
            int tl = r * 8 + tb;
            int t  = t0 + tl;
            if (t < TOUT)
                out[((size_t)b * TOUT + t) * 256 + (k0 + kl)] = tile[tl][kl];
        }
    }
}

// ---------------------------------------------------------------- launcher
extern "C" void kernel_launch(void* const* d_in, const int* in_sizes, int n_in,
                              void* d_out, int out_size, void* d_ws, size_t ws_size,
                              hipStream_t stream) {
    const float* X   = (const float*)d_in[0];   // (8, 514, 4000) f32
    const float* win = (const float*)d_in[1];   // (1024,) f32
    float* out = (float*)d_out;                 // (8, 3999*256) f32

    unsigned short* Whi = (unsigned short*)d_ws;                  // 512 KB
    unsigned short* Wlo = Whi + (size_t)NROWS * KDIM;             // 512 KB
    float* Y = (float*)((char*)d_ws + (size_t)2 * NROWS * KDIM * 2); // 65.5 MB

    build_w_kernel<<<dim3((NROWS * KDIM) / 256), 256, 0, stream>>>(Whi, Wlo);
    gemm_kernel<<<dim3((TDIM + BN - 1) / BN, NROWS / BM, BATCH), 256, 0, stream>>>(X, Whi, Wlo, Y);
    epilogue_kernel<<<dim3((TOUT + 31) / 32, 256 / 32, BATCH), 256, 0, stream>>>(Y, win, out);
}
